// DeformableTransformerEncoder_37254546325718
// MI455X (gfx1250) — matrix operand order repaired and checked by reading, not compiled, not run on hardware
//
#include <hip/hip_runtime.h>

// ---------------- static model config (from reference setup_inputs) ----------
#define BB    2
#define NN    13294      // total tokens
#define KQ    3323       // sparse queries
#define DD    256        // d_model
#define NHEAD 8
#define DHEAD 32
#define LLV   4          // levels
#define NPTS  4          // points per level
#define DFF_  1024
#define NLAY  6
#define N16   13296      // NN padded to 16
#define K16   3328       // KQ padded to 16

typedef __bf16 bf16_t;
typedef __attribute__((ext_vector_type(8)))  __bf16 v8bf;
typedef __attribute__((ext_vector_type(16))) __bf16 v16bf;
typedef __attribute__((ext_vector_type(8)))  float  v8f;

__device__ __forceinline__ bf16_t f2bf(float f) {
    unsigned u = __float_as_uint(f);
    u += 0x7fffu + ((u >> 16) & 1u);          // round-to-nearest-even
    unsigned short h = (unsigned short)(u >> 16);
    return __builtin_bit_cast(bf16_t, h);
}

#define CAT16 0,1,2,3,4,5,6,7,8,9,10,11,12,13,14,15

struct FragSet {
    v8bf a0l, a0h, a1l, a1h;   // two 16x32 A tiles (rows 0-15, 16-31)
    v8bf bl[4], bh[4];         // four 32x16 B tiles (cols 0..63)
};

// =====================================================================
// bf16 WMMA GEMM:  C[M x N] = A[M x K] * B[K x N] (+bias, relu)
// wave32: each wave -> 32 rows x 64 cols (8 WMMA tiles); block = 4 waves.
// Ping-pong (2x unrolled) K loop: load tile k+32/k+64 while computing k.
// K must be a multiple of 64 (holds: 256, 1024).
// A layout per ISA 7.12.2 (16-bit A 16x32); B per lane: n = lane%16, 16 k elems in A's k order.
// =====================================================================
__global__ void __launch_bounds__(128, 1)
gemm_bf16_wmma(const bf16_t* __restrict__ A, int lda,
               const bf16_t* __restrict__ Bw, int ldb,
               const float*  __restrict__ bias,
               float*  __restrict__ C,   int ldc,
               bf16_t* __restrict__ Cbf, int ldcb,
               int Mtiles32, int K, int relu)
{
    const int lane    = threadIdx.x & 31;
    const int wave    = threadIdx.x >> 5;
    const int rowTile = blockIdx.y * 4 + wave;     // 32-row tiles
    if (rowTile >= Mtiles32) return;               // wave-uniform, EXEC stays full
    const int row0 = rowTile << 5;
    const int col0 = blockIdx.x << 6;

    const int m  = lane & 15;
    const int kb = (lane >> 4) << 3;               // 0 or 8
    const bf16_t* __restrict__ ar0 = A + (size_t)(row0 + m) * lda + kb;
    const bf16_t* __restrict__ ar1 = ar0 + (size_t)16 * lda;
    const bf16_t* __restrict__ br  = Bw + (size_t)kb * ldb + col0 + m;
    const size_t  bstep = (size_t)32 * ldb;

    v8f acc[8] = {{},{},{},{},{},{},{},{}};

    auto loadFrag = [&](FragSet& f, int ka, const bf16_t* bp) {
        f.a0l = *(const v8bf*)(ar0 + ka);
        f.a0h = *(const v8bf*)(ar0 + ka + 16);
        f.a1l = *(const v8bf*)(ar1 + ka);
        f.a1h = *(const v8bf*)(ar1 + ka + 16);
        #pragma unroll
        for (int t = 0; t < 4; ++t) {
            #pragma unroll
            for (int i = 0; i < 8; ++i) {
                f.bl[t][i] = bp[(size_t)i * ldb + (t << 4)];
                f.bh[t][i] = bp[(size_t)(16 + i) * ldb + (t << 4)];
            }
        }
    };

    auto computeFrag = [&](const FragSet& f) {
        v16bf av0 = __builtin_shufflevector(f.a0l, f.a0h, CAT16);
        v16bf av1 = __builtin_shufflevector(f.a1l, f.a1h, CAT16);
        #pragma unroll
        for (int t = 0; t < 4; ++t) {
            v16bf bv = __builtin_shufflevector(f.bl[t], f.bh[t], CAT16);
            acc[t]     = __builtin_amdgcn_wmma_f32_16x16x32_bf16(
                             false, av0, false, bv, (short)0, acc[t],     false, false);
            acc[4 + t] = __builtin_amdgcn_wmma_f32_16x16x32_bf16(
                             false, av1, false, bv, (short)0, acc[4 + t], false, false);
        }
    };

    FragSet f0, f1;
    const bf16_t* bp = br;
    loadFrag(f0, 0, bp);

    int k0 = 0;
    for (; k0 + 64 < K; k0 += 64) {
        loadFrag(f1, k0 + 32, bp + bstep);       // prefetch tile k0+32
        computeFrag(f0);                         // compute tile k0
        loadFrag(f0, k0 + 64, bp + 2 * bstep);   // prefetch tile k0+64
        computeFrag(f1);                         // compute tile k0+32
        bp += 2 * bstep;
    }
    loadFrag(f1, k0 + 32, bp + bstep);           // last tile
    computeFrag(f0);
    computeFrag(f1);

    // ---- epilogue: bias / relu / store f32 and/or bf16 ----
    #pragma unroll
    for (int rt = 0; rt < 2; ++rt) {
        const int rbase = row0 + (rt << 4) + ((lane >> 4) << 3);
        #pragma unroll
        for (int t = 0; t < 4; ++t) {
            const int n  = col0 + (t << 4) + (lane & 15);
            const float bv = bias ? bias[n] : 0.f;
            #pragma unroll
            for (int r = 0; r < 8; ++r) {
                float v = acc[rt * 4 + t][r] + bv;
                if (relu) v = fmaxf(v, 0.f);
                if (C)   C  [(size_t)(rbase + r) * ldc  + n] = v;
                if (Cbf) Cbf[(size_t)(rbase + r) * ldcb + n] = f2bf(v);
            }
        }
    }
}

// ----------------------- small elementwise kernels --------------------------
__global__ void cvt_bf16(const float* __restrict__ in, bf16_t* __restrict__ out, int n) {
    int i = blockIdx.x * blockDim.x + threadIdx.x;
    if (i < n) out[i] = f2bf(in[i]);
}

__global__ void addcvt_bf16(const float* __restrict__ a, const float* __restrict__ b,
                            bf16_t* __restrict__ out, int n) {
    int i = blockIdx.x * blockDim.x + threadIdx.x;
    if (i < n) out[i] = f2bf(a[i] + b[i]);
}

// output [B,NN,D] (f32) -> padded [B*N16, D] bf16
__global__ void padcvt_out(const float* __restrict__ src, bf16_t* __restrict__ out) {
    int i = blockIdx.x * blockDim.x + threadIdx.x;
    if (i >= BB * N16 * DD) return;
    int col = i & (DD - 1);
    int row = i >> 8;
    int b   = row / N16;
    int r   = row - b * N16;
    float v = (r < NN) ? src[((size_t)b * NN + r) * DD + col] : 0.f;
    out[i] = f2bf(v);
}

// tgt/pos gather: tgt[(b*K16+k)*D+d] = src[(b*N + topk[b,k])*D+d]
__global__ void gather_tgt(const float* __restrict__ src, const float* __restrict__ pos,
                           const int* __restrict__ topk,
                           float* __restrict__ tgt, float* __restrict__ posq) {
    int i = blockIdx.x * blockDim.x + threadIdx.x;
    if (i >= BB * KQ * DD) return;
    int d = i & (DD - 1);
    int k = (i >> 8) % KQ;
    int b = i / (KQ * DD);
    int n = topk[b * KQ + k];
    size_t so = ((size_t)b * NN  + n) * DD + d;
    size_t to = ((size_t)b * K16 + k) * DD + d;
    tgt [to] = src[so];
    posq[to] = pos[so];
}

// per (b,k,h): softmax over 16 logits -> aw (in d_out), loc = ref + off/norm
__global__ void softmax_loc(const float* __restrict__ logits,
                            const float* __restrict__ offr,
                            const int*   __restrict__ topk,
                            float* __restrict__ locOut, float* __restrict__ awOut, int li)
{
    int i = blockIdx.x * blockDim.x + threadIdx.x;
    if (i >= BB * KQ * NHEAD) return;
    int h = i & 7;
    int k = (i >> 3) % KQ;
    int b = i / (KQ * NHEAD);

    const float* lg = logits + ((size_t)b * K16 + k) * 128 + h * 16;
    float mx = lg[0];
    #pragma unroll
    for (int j = 1; j < 16; ++j) mx = fmaxf(mx, lg[j]);
    float e[16], s = 0.f;
    #pragma unroll
    for (int j = 0; j < 16; ++j) { e[j] = expf(lg[j] - mx); s += e[j]; }
    float inv = 1.f / s;
    size_t awBase = (((size_t)b * NLAY + li) * KQ + k) * 128 + h * 16;
    #pragma unroll
    for (int j = 0; j < 16; ++j) awOut[awBase + j] = e[j] * inv;

    // reference point for token n
    int n = topk[b * KQ + k];
    int l0  = (n < 10000) ? 0 : (n < 12500) ? 1 : (n < 13125) ? 2 : 3;
    const int Ws[4] = {100, 50, 25, 13};
    const int Hs[4] = {100, 50, 25, 13};
    const int St[4] = {0, 10000, 12500, 13125};
    int idx = n - St[l0];
    int yy  = idx / Ws[l0];
    int xx  = idx - yy * Ws[l0];
    float rx = (xx + 0.5f) / (float)Ws[l0];
    float ry = (yy + 0.5f) / (float)Hs[l0];

    size_t offBase = ((size_t)b * K16 + k) * 256 + h * 32;
    size_t locBase = (((size_t)b * NLAY + li) * KQ + k) * 256 + h * 32;
    #pragma unroll
    for (int l = 0; l < 4; ++l) {
        #pragma unroll
        for (int p = 0; p < 4; ++p) {
            float ox = offr[offBase + (l * 4 + p) * 2 + 0];
            float oy = offr[offBase + (l * 4 + p) * 2 + 1];
            locOut[locBase + (l * 4 + p) * 2 + 0] = rx + ox / (float)Ws[l];
            locOut[locBase + (l * 4 + p) * 2 + 1] = ry + oy / (float)Hs[l];
        }
    }
}

// multi-scale deformable attention: one wave per (b,k,h); lane = channel (Dh=32)
__global__ void msda(const float* __restrict__ value,   // [B*N16, 256]
                     const float* __restrict__ loc,     // d_out loc base
                     const float* __restrict__ aw,      // d_out aw base
                     float* __restrict__ out, int li)   // [B*K16, 256]
{
    int gw   = (blockIdx.x * blockDim.x + threadIdx.x) >> 5;
    int lane = threadIdx.x & 31;
    if (gw >= BB * KQ * NHEAD) return;
    int h = gw & 7;
    int k = (gw >> 3) % KQ;
    int b = gw / (KQ * NHEAD);

    size_t locBase = (((size_t)b * NLAY + li) * KQ + k) * 256 + h * 32;
    size_t awBase  = (((size_t)b * NLAY + li) * KQ + k) * 128 + h * 16;
    int ccol = h * 32 + lane;

    const int Ws[4] = {100, 50, 25, 13};
    const int Hs[4] = {100, 50, 25, 13};
    const int St[4] = {0, 10000, 12500, 13125};

    float acc = 0.f;
    #pragma unroll
    for (int l = 0; l < 4; ++l) {
        const float* vb = value + ((size_t)b * N16 + St[l]) * 256;
        const int W = Ws[l], H = Hs[l];
        #pragma unroll
        for (int p = 0; p < 4; ++p) {
            float lx = loc[locBase + (l * 4 + p) * 2 + 0];
            float ly = loc[locBase + (l * 4 + p) * 2 + 1];
            float a  = aw [awBase + l * 4 + p];
            float x = lx * (float)W - 0.5f;
            float y = ly * (float)H - 0.5f;
            float x0f = floorf(x), y0f = floorf(y);
            int   x0 = (int)x0f,  y0 = (int)y0f;
            float fx = x - x0f,   fy = y - y0f;
            float w00 = (1.f - fx) * (1.f - fy);
            float w10 = fx * (1.f - fy);
            float w01 = (1.f - fx) * fy;
            float w11 = fx * fy;
            if (x0   >= 0 && x0   < W && y0   >= 0 && y0   < H)
                acc += a * w00 * vb[((size_t)(y0 * W + x0)) * 256 + ccol];
            if (x0+1 >= 0 && x0+1 < W && y0   >= 0 && y0   < H)
                acc += a * w10 * vb[((size_t)(y0 * W + x0 + 1)) * 256 + ccol];
            if (x0   >= 0 && x0   < W && y0+1 >= 0 && y0+1 < H)
                acc += a * w01 * vb[((size_t)((y0 + 1) * W + x0)) * 256 + ccol];
            if (x0+1 >= 0 && x0+1 < W && y0+1 >= 0 && y0+1 < H)
                acc += a * w11 * vb[((size_t)((y0 + 1) * W + x0 + 1)) * 256 + ccol];
        }
    }
    out[((size_t)b * K16 + k) * 256 + ccol] = acc;
}

// x = LayerNorm(A + Badd); one 256-thread block per (b,k) row
__global__ void add_ln(const float* __restrict__ A, const float* __restrict__ Badd,
                       const float* __restrict__ g, const float* __restrict__ be,
                       float* __restrict__ outF, bf16_t* __restrict__ outBf)
{
    __shared__ float sh[256];
    int row = blockIdx.x;
    int b = row / KQ, k = row - b * KQ;
    size_t base = ((size_t)b * K16 + k) * 256;
    int t = threadIdx.x;
    float v = A[base + t] + Badd[base + t];
    sh[t] = v; __syncthreads();
    for (int s = 128; s > 0; s >>= 1) { if (t < s) sh[t] += sh[t + s]; __syncthreads(); }
    float mean = sh[0] * (1.f / 256.f); __syncthreads();
    float d = v - mean;
    sh[t] = d * d; __syncthreads();
    for (int s = 128; s > 0; s >>= 1) { if (t < s) sh[t] += sh[t + s]; __syncthreads(); }
    float var = sh[0] * (1.f / 256.f);
    float y = d * rsqrtf(var + 1e-5f) * g[t] + be[t];
    if (outF)  outF [base + t] = y;
    if (outBf) outBf[base + t] = f2bf(y);
}

// x2 = LN(A + Badd); tgt <- x2; ragged scatter into output[b, topk[b,k]]
__global__ void add_ln_scatter(const float* __restrict__ A, const float* __restrict__ Badd,
                               const float* __restrict__ g, const float* __restrict__ be,
                               float* __restrict__ tgtOut, float* __restrict__ outGlobal,
                               const int* __restrict__ topk, const int* __restrict__ stn)
{
    __shared__ float sh[256];
    int row = blockIdx.x;
    int b = row / KQ, k = row - b * KQ;
    size_t base = ((size_t)b * K16 + k) * 256;
    int t = threadIdx.x;
    float v = A[base + t] + Badd[base + t];
    sh[t] = v; __syncthreads();
    for (int s = 128; s > 0; s >>= 1) { if (t < s) sh[t] += sh[t + s]; __syncthreads(); }
    float mean = sh[0] * (1.f / 256.f); __syncthreads();
    float d = v - mean;
    sh[t] = d * d; __syncthreads();
    for (int s = 128; s > 0; s >>= 1) { if (t < s) sh[t] += sh[t + s]; __syncthreads(); }
    float var = sh[0] * (1.f / 256.f);
    float y = d * rsqrtf(var + 1e-5f) * g[t] + be[t];
    tgtOut[base + t] = y;
    if (k < stn[b]) {
        int n = topk[b * KQ + k];
        outGlobal[((size_t)b * NN + n) * 256 + t] = y;
    }
}

// ============================ host driver ====================================
extern "C" void kernel_launch(void* const* d_in, const int* in_sizes, int n_in,
                              void* d_out, int out_size, void* d_ws, size_t ws_size,
                              hipStream_t stream)
{
    (void)in_sizes; (void)n_in; (void)out_size; (void)ws_size;
    const float* src  = (const float*)d_in[0];
    const float* pos  = (const float*)d_in[1];
    const int*   topk = (const int*)d_in[4];
    const int*   stn  = (const int*)d_in[5];

    float* outBuf = (float*)d_out;                              // [B,N,D]
    float* locOut = outBuf + (size_t)BB * NN * DD;              // [B,NL,K,256]
    float* awOut  = locOut + (size_t)BB * NLAY * KQ * 256;      // [B,NL,K,128]

    // --- bump-allocate workspace ---
    char* w = (char*)d_ws;
    auto alloc = [&](size_t bytes) -> void* {
        void* p = (void*)w; w += (bytes + 255) & ~(size_t)255; return p;
    };
    float*  tgt    = (float*) alloc((size_t)BB * K16 * DD  * 4);
    float*  posq   = (float*) alloc((size_t)BB * K16 * DD  * 4);
    float*  value  = (float*) alloc((size_t)BB * N16 * DD  * 4);
    float*  offr   = (float*) alloc((size_t)BB * K16 * 256 * 4);
    float*  logits = (float*) alloc((size_t)BB * K16 * 128 * 4);
    float*  attnO  = (float*) alloc((size_t)BB * K16 * DD  * 4);
    float*  attnP  = (float*) alloc((size_t)BB * K16 * DD  * 4);
    float*  xbuf   = (float*) alloc((size_t)BB * K16 * DD  * 4);
    float*  ff2    = (float*) alloc((size_t)BB * K16 * DD  * 4);
    bf16_t* qbf    = (bf16_t*)alloc((size_t)BB * K16 * DD   * 2);
    bf16_t* outbf  = (bf16_t*)alloc((size_t)BB * N16 * DD   * 2);
    bf16_t* aobf   = (bf16_t*)alloc((size_t)BB * K16 * DD   * 2);
    bf16_t* xbf    = (bf16_t*)alloc((size_t)BB * K16 * DD   * 2);
    bf16_t* ff1bf  = (bf16_t*)alloc((size_t)BB * K16 * DFF_ * 2);
    bf16_t* wv     = (bf16_t*)alloc(65536  * 2);
    bf16_t* woff   = (bf16_t*)alloc(65536  * 2);
    bf16_t* wattn  = (bf16_t*)alloc(32768  * 2);
    bf16_t* wout   = (bf16_t*)alloc(65536  * 2);
    bf16_t* wff1   = (bf16_t*)alloc(262144 * 2);
    bf16_t* wff2   = (bf16_t*)alloc(262144 * 2);

    // output = src
    hipMemcpyAsync(outBuf, src, (size_t)BB * NN * DD * 4, hipMemcpyDeviceToDevice, stream);

    { int n = BB * KQ * DD;
      gather_tgt<<<(n + 255) / 256, 256, 0, stream>>>(src, pos, topk, tgt, posq); }

    const int MtK = BB * K16 / 32;   // 208 query 32-row tiles
    const int MtN = BB * N16 / 32;   // 831 token 32-row tiles

    for (int li = 0; li < NLAY; ++li) {
        const float* const* P = (const float* const*)(d_in + 6 + li * 16);
        const float *W_v = P[0],  *b_v = P[1],  *W_off = P[2], *b_off = P[3];
        const float *W_at = P[4], *b_at = P[5], *W_o = P[6],   *b_o = P[7];
        const float *ln1g = P[8], *ln1b = P[9];
        const float *W_f1 = P[10],*b_f1 = P[11],*W_f2 = P[12], *b_f2 = P[13];
        const float *ln2g = P[14],*ln2b = P[15];

        // weights -> bf16
        cvt_bf16<<<256, 256, 0, stream>>>(W_v,  wv,    65536);
        cvt_bf16<<<256, 256, 0, stream>>>(W_off,woff,  65536);
        cvt_bf16<<<128, 256, 0, stream>>>(W_at, wattn, 32768);
        cvt_bf16<<<256, 256, 0, stream>>>(W_o,  wout,  65536);
        cvt_bf16<<<1024,256, 0, stream>>>(W_f1, wff1,  262144);
        cvt_bf16<<<1024,256, 0, stream>>>(W_f2, wff2,  262144);

        // q = tgt + pos_q -> bf16 ; output -> padded bf16
        { int n = BB * K16 * DD;
          addcvt_bf16<<<(n + 255) / 256, 256, 0, stream>>>(tgt, posq, qbf, n); }
        { int n = BB * N16 * DD;
          padcvt_out<<<(n + 255) / 256, 256, 0, stream>>>(outBuf, outbf); }

        // value = output @ W_v + b_v
        { dim3 g(DD / 64, (MtN + 3) / 4);
          gemm_bf16_wmma<<<g, 128, 0, stream>>>(outbf, DD, wv, DD, b_v,
                                                value, DD, nullptr, 0, MtN, DD, 0); }
        // offsets, attention logits
        { dim3 g(256 / 64, (MtK + 3) / 4);
          gemm_bf16_wmma<<<g, 128, 0, stream>>>(qbf, DD, woff, 256, b_off,
                                                offr, 256, nullptr, 0, MtK, DD, 0); }
        { dim3 g(128 / 64, (MtK + 3) / 4);
          gemm_bf16_wmma<<<g, 128, 0, stream>>>(qbf, DD, wattn, 128, b_at,
                                                logits, 128, nullptr, 0, MtK, DD, 0); }

        { int n = BB * KQ * NHEAD;
          softmax_loc<<<(n + 255) / 256, 256, 0, stream>>>(logits, offr, topk,
                                                           locOut, awOut, li); }
        { int nw = BB * KQ * NHEAD;
          msda<<<(nw + 3) / 4, 128, 0, stream>>>(value, locOut, awOut, attnO, li); }

        // attn @ W_out + b_out
        { int n = BB * K16 * DD;
          cvt_bf16<<<(n + 255) / 256, 256, 0, stream>>>(attnO, aobf, n); }
        { dim3 g(DD / 64, (MtK + 3) / 4);
          gemm_bf16_wmma<<<g, 128, 0, stream>>>(aobf, DD, wout, DD, b_o,
                                                attnP, DD, nullptr, 0, MtK, DD, 0); }

        // x = LN(tgt + attn)
        add_ln<<<BB * KQ, 256, 0, stream>>>(tgt, attnP, ln1g, ln1b, xbuf, xbf);

        // FFN: relu(x@W1+b1)@W2+b2
        { dim3 g(DFF_ / 64, (MtK + 3) / 4);
          gemm_bf16_wmma<<<g, 128, 0, stream>>>(xbf, DD, wff1, DFF_, b_f1,
                                                nullptr, 0, ff1bf, DFF_, MtK, DD, 1); }
        { dim3 g(DD / 64, (MtK + 3) / 4);
          gemm_bf16_wmma<<<g, 128, 0, stream>>>(ff1bf, DFF_, wff2, DD, b_f2,
                                                ff2, DD, nullptr, 0, MtK, DFF_, 0); }

        // tgt = LN(x + ff); ragged scatter into output
        add_ln_scatter<<<BB * KQ, 256, 0, stream>>>(xbuf, ff2, ln2g, ln2b,
                                                    tgt, outBuf, topk, stn);
    }
}
